// EQGATLocalGNN_83700322665130
// MI455X (gfx1250) — matrix operand order repaired and checked by reading, not compiled, not run on hardware
//
#include <hip/hip_runtime.h>
#include <hip/hip_bf16.h>
#include <cmath>

typedef __bf16 bf16_t;
typedef __attribute__((ext_vector_type(16))) __bf16 v16bf;
typedef __attribute__((ext_vector_type(8)))  float  v8f;

#define N_NODES 32768
#define E_EDGES 524288
#define S_DIM   128
#define V_DIM   32
#define ED_DIM  32
#define G_GRAPH 256
#define EPSV    1e-6f

// ---------------- WMMA helpers (CDNA5 16x16x32 bf16, fp32 accum) ----------------

__device__ __forceinline__ v8f wmma_bf16(v16bf a, v16bf b, v8f c) {
  return __builtin_amdgcn_wmma_f32_16x16x32_bf16(false, a, false, b, (short)0, c, false, false);
}

// A-matrix 16x32 bf16 fragment. base points at (row 0 of tile, kk). ld in elements.
// lane 0-15: M=lane, K halves {0..7},{16..23}; lane 16-31: M=lane-16, K {8..15},{24..31}.
__device__ __forceinline__ v16bf load_A_frag(const bf16_t* base, int ld, int lane) {
  const int row = lane & 15;
  const int kh  = (lane >> 4) << 3;     // 0 or 8
  const bf16_t* p = base + row * ld + kh;
  v16bf f;
#pragma unroll
  for (int i = 0; i < 8; ++i) f[i] = p[i];
#pragma unroll
  for (int i = 0; i < 8; ++i) f[i + 8] = p[16 + i];
  return f;
}

// B-matrix 32x16 bf16 fragment from a [dout][ldk] (K-contiguous, transposed) weight.
// lane 0-15: col N=lane, K {0..7},{16..23}; lane 16-31: col N=lane-16, K {8..15},{24..31}.
__device__ __forceinline__ v16bf load_B_frag(const bf16_t* WT, int ldk, int kk, int nn, int lane) {
  const bf16_t* p = WT + (size_t)(nn + (lane & 15)) * ldk + kk + ((lane >> 4) << 3);
  v16bf f;
#pragma unroll
  for (int i = 0; i < 8; ++i) f[i] = p[i];
#pragma unroll
  for (int i = 0; i < 8; ++i) f[i + 8] = p[16 + i];
  return f;
}

__device__ __forceinline__ float silu_f(float x) { return x / (1.f + __expf(-x)); }

// ---------------- weight repack: f32 [din][dout] -> bf16 [dout][dinp] ----------------

__global__ void k_w2bf(const float* __restrict__ W, bf16_t* __restrict__ out,
                       int din, int dout, int dinp) {
  const int col = blockIdx.x;
  for (int k = threadIdx.x; k < dinp; k += blockDim.x)
    out[(size_t)col * dinp + k] = (k < din) ? (bf16_t)W[(size_t)k * dout + col] : (bf16_t)0.f;
}

// ---------------- degree / graph counts (computed once) ----------------

__global__ void k_counts(const int* __restrict__ dstI, float* __restrict__ deg,
                         const int* __restrict__ batch, float* __restrict__ gcnt,
                         int E, int N) {
  const int i = blockIdx.x * blockDim.x + threadIdx.x;
  if (i < E) atomicAdd(&deg[dstI[i]], 1.f);
  if (i < N) atomicAdd(&gcnt[batch[i]], 1.f);
}

// ---------------- per-node vector-norm reduction into graph sums ----------------

__global__ __launch_bounds__(128) void k_vnorm(const float* __restrict__ v_cur,
                                               const int* __restrict__ batch,
                                               float* __restrict__ gsum) {
  const int node = blockIdx.x * 4 + (threadIdx.x >> 5);
  const int lane = threadIdx.x & 31;
  const size_t vo = (size_t)node * 96;
  float s = 0.f;
#pragma unroll
  for (int dim = 0; dim < 3; ++dim) { float t = v_cur[vo + dim * 32 + lane]; s += t * t; }
#pragma unroll
  for (int m = 16; m >= 1; m >>= 1) s += __shfl_xor(s, m, 32);
  if (lane == 0) atomicAdd(&gsum[batch[node]], s * (1.f / 32.f));
}

// ---------------- node prep: LayerNorm(s), v scaling, vb = v @ Wvec ----------------

__global__ __launch_bounds__(128) void k_node_prep(
    float* __restrict__ s_cur, float* __restrict__ v_cur,
    bf16_t* __restrict__ s_bf, float* __restrict__ vbm,
    const int* __restrict__ batch, const float* __restrict__ gsum,
    const float* __restrict__ gcnt,
    const float* __restrict__ ln_w, const float* __restrict__ ln_b,
    const float* __restrict__ Wvec, int has_vec) {
  const int node = blockIdx.x * 4 + (threadIdx.x >> 5);
  const int lane = threadIdx.x & 31;
  const size_t so = (size_t)node * S_DIM;

  float x[4]; float sum = 0.f;
#pragma unroll
  for (int i = 0; i < 4; ++i) { x[i] = s_cur[so + lane + 32 * i]; sum += x[i]; }
#pragma unroll
  for (int m = 16; m >= 1; m >>= 1) sum += __shfl_xor(sum, m, 32);
  const float mu = sum * (1.f / 128.f);
  float vs = 0.f;
#pragma unroll
  for (int i = 0; i < 4; ++i) { x[i] -= mu; vs += x[i] * x[i]; }
#pragma unroll
  for (int m = 16; m >= 1; m >>= 1) vs += __shfl_xor(vs, m, 32);
  const float rstd = rsqrtf(vs * (1.f / 128.f) + EPSV);
#pragma unroll
  for (int i = 0; i < 4; ++i) {
    const int c = lane + 32 * i;
    const float y = x[i] * rstd * ln_w[c] + ln_b[c];
    s_cur[so + c] = y;
    s_bf[so + c]  = (bf16_t)y;
  }

  const int g = batch[node];
  const float vmean = gsum[g] / fmaxf(gcnt[g], 1.f);
  const float sc = rsqrtf(vmean + EPSV);
  const size_t vo = (size_t)node * 96;
  float vd[3];
#pragma unroll
  for (int dim = 0; dim < 3; ++dim) {
    vd[dim] = v_cur[vo + dim * 32 + lane] * sc;
    v_cur[vo + dim * 32 + lane] = vd[dim];
  }
  if (has_vec) {
    float acc[3] = {0.f, 0.f, 0.f};
    for (int k = 0; k < 32; ++k) {
      const float w = Wvec[k * 32 + lane];
#pragma unroll
      for (int dim = 0; dim < 3; ++dim) acc[dim] += __shfl(vd[dim], k, 32) * w;
    }
#pragma unroll
    for (int dim = 0; dim < 3; ++dim) vbm[vo + dim * 32 + lane] = acc[dim];
  }
}

// ---------------- sb = s_ln @ Ws + bs   (N x 128, K=128, WMMA) ----------------

__global__ __launch_bounds__(128) void k_sb(const bf16_t* __restrict__ s_bf,
                                            const bf16_t* __restrict__ WsT,
                                            const float* __restrict__ bs,
                                            float* __restrict__ sb) {
  const int lane = threadIdx.x & 31, wv = threadIdx.x >> 5;
  const int row0 = blockIdx.x * 64 + wv * 16;
  v8f acc[8];
#pragma unroll
  for (int t = 0; t < 8; ++t) {
    const float b = bs[t * 16 + (lane & 15)];
#pragma unroll
    for (int r = 0; r < 8; ++r) acc[t][r] = b;
  }
  for (int kk = 0; kk < 128; kk += 32) {
    const v16bf a = load_A_frag(s_bf + (size_t)row0 * S_DIM + kk, S_DIM, lane);
#pragma unroll
    for (int t = 0; t < 8; ++t) {
      const v16bf b = load_B_frag(WsT, 128, kk, t * 16, lane);
      acc[t] = wmma_bf16(a, b, acc[t]);
    }
  }
  const int mofs = (lane >> 4) * 8;
#pragma unroll
  for (int t = 0; t < 8; ++t)
#pragma unroll
    for (int r = 0; r < 8; ++r)
      sb[(size_t)(row0 + mofs + r) * S_DIM + t * 16 + (lane & 15)] = acc[t][r];
}

// ---------------- fused edge kernel ----------------
// 64 edges/block, 4 waves; wave w owns edge M-tile w across all N-tiles.
// feat = [s[dst] | s[src] | d | a | e | pad] (K=320 bf16 in LDS, vectorized gather)
// H = silu(feat @ We1 + be1)  (64x128, LDS bf16)
// OUT = H @ We2 + be2 applied DIRECTLY from WMMA D fragments:
//   gate cols (n<128):    s_acc[dst] += gate * sb[src]          (atomic f32)
//   v0 cols (128<=n<160): v_acc[dst] += r*v0 + v1*vb[src], v1 = same lane, tile t+2

#define EB   64
#define FLD  328
#define HLD  136

template <int NT2>   // 10 (nout=160, layer 0) or 12 (nout=192, Wvec layers)
__global__ __launch_bounds__(128) void k_edge(
    const bf16_t* __restrict__ s_bf, const float* __restrict__ sb,
    const float* __restrict__ vbm,
    const int* __restrict__ srcI, const int* __restrict__ dstI,
    const float* __restrict__ dA, const float* __restrict__ aA,
    const float* __restrict__ eA, const float* __restrict__ rA,
    const bf16_t* __restrict__ We1T, const float* __restrict__ be1,
    const bf16_t* __restrict__ We2T, const float* __restrict__ be2,
    float* __restrict__ s_acc, float* __restrict__ v_acc) {
  __shared__ __align__(16) bf16_t feat[EB * FLD];
  __shared__ __align__(16) bf16_t Hs[EB * HLD];

  const int tid = threadIdx.x, lane = tid & 31, wv = tid >> 5;
  const int e0 = blockIdx.x * EB;
  const int nc = lane & 15;
  const int mofs = (lane >> 4) * 8;
  const int mm = wv * 16;

  // gather s[dst]|s[src] as 16B chunks: 64 edges * 32 chunks
  for (int i = tid; i < EB * 32; i += 128) {
    const int ed = i >> 5;
    const int ch = i & 31;
    const int ge = e0 + ed;
    const int half = ch >> 4;                 // 0: dst row, 1: src row
    const int cc = (ch & 15) << 3;            // bf16 offset within row
    const int node = half ? srcI[ge] : dstI[ge];
    *(uint4*)&feat[ed * FLD + half * S_DIM + cc] =
        *(const uint4*)(s_bf + (size_t)node * S_DIM + cc);
  }
  // tail cols 256..319: d, a, e, zero pad
  for (int i = tid; i < EB * 64; i += 128) {
    const int ed = i >> 6;
    const int c = 256 + (i & 63);
    const int ge = e0 + ed;
    float val;
    if      (c == 256) val = dA[ge];
    else if (c == 257) val = aA[ge];
    else if (c < 290)  val = eA[(size_t)ge * ED_DIM + (c - 258)];
    else val = 0.f;
    feat[ed * FLD + c] = (bf16_t)val;
  }
  __syncthreads();

  { // stage 1: K=320 -> 128, silu; wave covers its 16 edges x all 128 cols
    v8f acc[8];
#pragma unroll
    for (int t = 0; t < 8; ++t) {
      const float b = be1[t * 16 + nc];
#pragma unroll
      for (int r = 0; r < 8; ++r) acc[t][r] = b;
    }
    for (int kk = 0; kk < 320; kk += 32) {
      const v16bf a = load_A_frag(&feat[mm * FLD + kk], FLD, lane);
#pragma unroll
      for (int t = 0; t < 8; ++t) {
        const v16bf b = load_B_frag(We1T, 320, kk, t * 16, lane);
        acc[t] = wmma_bf16(a, b, acc[t]);
      }
    }
#pragma unroll
    for (int t = 0; t < 8; ++t)
#pragma unroll
      for (int r = 0; r < 8; ++r)
        Hs[(mm + mofs + r) * HLD + t * 16 + nc] = (bf16_t)silu_f(acc[t][r]);
  }
  __syncthreads();

  { // stage 2: K=128 -> nout, then scatter straight from D fragments
    v8f acc[NT2];
#pragma unroll
    for (int t = 0; t < NT2; ++t) {
      const float b = be2[t * 16 + nc];
#pragma unroll
      for (int r = 0; r < 8; ++r) acc[t][r] = b;
    }
    for (int kk = 0; kk < 128; kk += 32) {
      const v16bf a = load_A_frag(&Hs[mm * HLD + kk], HLD, lane);
#pragma unroll
      for (int t = 0; t < NT2; ++t) {
        const v16bf b = load_B_frag(We2T, 128, kk, t * 16, lane);
        acc[t] = wmma_bf16(a, b, acc[t]);
      }
    }

    int sn8[8], dn8[8];
#pragma unroll
    for (int r = 0; r < 8; ++r) {
      const int ge = e0 + mm + mofs + r;
      sn8[r] = srcI[ge];
      dn8[r] = dstI[ge];
    }
#pragma unroll
    for (int t = 0; t < NT2; ++t) {
      if (t < 8) {                       // gate columns n = t*16+nc < 128
        const int n = t * 16 + nc;
#pragma unroll
        for (int r = 0; r < 8; ++r)
          atomicAdd(&s_acc[(size_t)dn8[r] * S_DIM + n],
                    acc[t][r] * sb[(size_t)sn8[r] * S_DIM + n]);
      } else if (t < 10) {               // v0 columns, wcol in [0,32)
        const int wcol = (t - 8) * 16 + nc;
#pragma unroll
        for (int r = 0; r < 8; ++r) {
          const int ge = e0 + mm + mofs + r;
          const float v0 = acc[t][r];
          const float v1 = (NT2 > 10) ? acc[(t + 2 < NT2) ? t + 2 : t][r] : 0.f;
#pragma unroll
          for (int dim = 0; dim < 3; ++dim) {
            float nv = rA[(size_t)ge * 3 + dim] * v0;
            if (NT2 > 10) nv += v1 * vbm[(size_t)sn8[r] * 96 + dim * 32 + wcol];
            atomicAdd(&v_acc[(size_t)dn8[r] * 96 + dim * 32 + wcol], nv);
          }
        }
      }
      // t >= 10: v1 columns, consumed by the v0 path above
    }
  }
}

// ---------------- node update: s += s_acc; v += v_acc/deg; vv = v@Wv0; sin ----------------

__global__ __launch_bounds__(128) void k_update(
    float* __restrict__ s_cur, float* __restrict__ v_cur,
    const float* __restrict__ s_acc, const float* __restrict__ v_acc,
    const float* __restrict__ deg, const float* __restrict__ Wv0,
    float* __restrict__ vv2, bf16_t* __restrict__ sinb) {
  const int node = blockIdx.x * 4 + (threadIdx.x >> 5);
  const int lane = threadIdx.x & 31;
  const size_t so = (size_t)node * S_DIM;
  float sv[4];
#pragma unroll
  for (int i = 0; i < 4; ++i) {
    const int c = lane + 32 * i;
    sv[i] = s_cur[so + c] + s_acc[so + c];
    s_cur[so + c] = sv[i];
  }
  const float inv = 1.f / fmaxf(deg[node], 1.f);
  const size_t vo = (size_t)node * 96;
  float vd[3];
#pragma unroll
  for (int dim = 0; dim < 3; ++dim) {
    vd[dim] = v_cur[vo + dim * 32 + lane] + v_acc[vo + dim * 32 + lane] * inv;
    v_cur[vo + dim * 32 + lane] = vd[dim];
  }
  float va[3] = {0.f, 0.f, 0.f}, vb2[3] = {0.f, 0.f, 0.f};
  for (int k = 0; k < 32; ++k) {
    const float w0 = Wv0[k * 64 + lane];
    const float w1 = Wv0[k * 64 + 32 + lane];
#pragma unroll
    for (int dim = 0; dim < 3; ++dim) {
      const float vk = __shfl(vd[dim], k, 32);
      va[dim]  += vk * w0;
      vb2[dim] += vk * w1;
    }
  }
  const float nrm = sqrtf(va[0] * va[0] + va[1] * va[1] + va[2] * va[2] + EPSV);
#pragma unroll
  for (int dim = 0; dim < 3; ++dim) vv2[vo + dim * 32 + lane] = vb2[dim];
  const size_t io = (size_t)node * 160;
#pragma unroll
  for (int i = 0; i < 4; ++i) sinb[io + lane + 32 * i] = (bf16_t)sv[i];
  sinb[io + 128 + lane] = (bf16_t)nrm;
}

// ---------------- update MLP: apply hu straight from D fragments ----------------

__device__ __forceinline__ void apply_out(float val, int gm, int n,
                                          float* s_cur, float* v_cur, const float* vv2) {
  if (n < 32) {
#pragma unroll
    for (int dim = 0; dim < 3; ++dim) {
      const size_t o = (size_t)gm * 96 + dim * 32 + n;
      v_cur[o] += val * vv2[o];
    }
  } else {
    s_cur[(size_t)gm * S_DIM + (n - 32)] += val;
  }
}

__global__ __launch_bounds__(128) void k_upmlp(
    const bf16_t* __restrict__ sinb,
    const bf16_t* __restrict__ Wu1T, const float* __restrict__ bu1,
    const bf16_t* __restrict__ Wu2T, const float* __restrict__ bu2,
    float* __restrict__ s_cur, float* __restrict__ v_cur,
    const float* __restrict__ vv2) {
  __shared__ __align__(16) bf16_t Hs[64 * HLD];
  const int lane = threadIdx.x & 31, wv = threadIdx.x >> 5;
  const int row0 = blockIdx.x * 64 + wv * 16;
  const int lrow = wv * 16;

  { // stage 1: K=160 -> 128, silu
    v8f acc[8];
#pragma unroll
    for (int t = 0; t < 8; ++t) {
      const float b = bu1[t * 16 + (lane & 15)];
#pragma unroll
      for (int r = 0; r < 8; ++r) acc[t][r] = b;
    }
    for (int kk = 0; kk < 160; kk += 32) {
      const v16bf a = load_A_frag(sinb + (size_t)row0 * 160 + kk, 160, lane);
#pragma unroll
      for (int t = 0; t < 8; ++t) {
        const v16bf b = load_B_frag(Wu1T, 160, kk, t * 16, lane);
        acc[t] = wmma_bf16(a, b, acc[t]);
      }
    }
    const int mofs = (lane >> 4) * 8;
#pragma unroll
    for (int t = 0; t < 8; ++t)
#pragma unroll
      for (int r = 0; r < 8; ++r)
        Hs[(lrow + mofs + r) * HLD + t * 16 + (lane & 15)] = (bf16_t)silu_f(acc[t][r]);
  }
  __syncthreads();

  { // stage 2: K=128 -> 160, apply
    v8f acc[10];
#pragma unroll
    for (int t = 0; t < 10; ++t) {
      const float b = bu2[t * 16 + (lane & 15)];
#pragma unroll
      for (int r = 0; r < 8; ++r) acc[t][r] = b;
    }
    for (int kk = 0; kk < 128; kk += 32) {
      const v16bf a = load_A_frag(&Hs[lrow * HLD + kk], HLD, lane);
#pragma unroll
      for (int t = 0; t < 10; ++t) {
        const v16bf b = load_B_frag(Wu2T, 128, kk, t * 16, lane);
        acc[t] = wmma_bf16(a, b, acc[t]);
      }
    }
    const int mofs = (lane >> 4) * 8;
#pragma unroll
    for (int t = 0; t < 10; ++t) {
      const int n = t * 16 + (lane & 15);
#pragma unroll
      for (int r = 0; r < 8; ++r)
        apply_out(acc[t][r], row0 + mofs + r, n, s_cur, v_cur, vv2);
    }
  }
}

__global__ __launch_bounds__(128) void k_uplast(
    const bf16_t* __restrict__ sinb,
    const bf16_t* __restrict__ Wu1T, const float* __restrict__ bu1,
    float* __restrict__ s_cur, float* __restrict__ v_cur,
    const float* __restrict__ vv2) {
  const int lane = threadIdx.x & 31, wv = threadIdx.x >> 5;
  const int row0 = blockIdx.x * 64 + wv * 16;
  v8f acc[10];
#pragma unroll
  for (int t = 0; t < 10; ++t) {
    const float b = bu1[t * 16 + (lane & 15)];
#pragma unroll
    for (int r = 0; r < 8; ++r) acc[t][r] = b;
  }
  for (int kk = 0; kk < 160; kk += 32) {
    const v16bf a = load_A_frag(sinb + (size_t)row0 * 160 + kk, 160, lane);
#pragma unroll
    for (int t = 0; t < 10; ++t) {
      const v16bf b = load_B_frag(Wu1T, 160, kk, t * 16, lane);
      acc[t] = wmma_bf16(a, b, acc[t]);
    }
  }
  const int mofs = (lane >> 4) * 8;
#pragma unroll
  for (int t = 0; t < 10; ++t) {
    const int n = t * 16 + (lane & 15);
#pragma unroll
    for (int r = 0; r < 8; ++r)
      apply_out(acc[t][r], row0 + mofs + r, n, s_cur, v_cur, vv2);
  }
}

// ---------------- host orchestration ----------------

extern "C" void kernel_launch(void* const* d_in, const int* in_sizes, int n_in,
                              void* d_out, int out_size, void* d_ws, size_t ws_size,
                              hipStream_t stream) {
  (void)in_sizes; (void)n_in; (void)out_size; (void)ws_size;

  const float* s_in  = (const float*)d_in[0];
  const float* v_in  = (const float*)d_in[1];
  // d_in[2] = p (unused by reference math)
  const int*   eidx  = (const int*)d_in[3];
  const float* dA    = (const float*)d_in[4];
  const float* aA    = (const float*)d_in[5];
  const float* rA    = (const float*)d_in[6];
  const float* eA    = (const float*)d_in[7];
  const int*   batch = (const int*)d_in[8];

  const int* srcI = eidx;
  const int* dstI = eidx + E_EDGES;

  char* w = (char*)d_ws;
  auto carve = [&](size_t bytes) -> void* {
    void* p = (void*)w;
    w += (bytes + 255) & ~(size_t)255;
    return p;
  };
  float*  s_cur = (float*)carve((size_t)N_NODES * 128 * 4);
  float*  v_cur = (float*)carve((size_t)N_NODES * 96 * 4);
  bf16_t* s_bf  = (bf16_t*)carve((size_t)N_NODES * 128 * 2);
  float*  sb    = (float*)carve((size_t)N_NODES * 128 * 4);
  float*  vbm   = (float*)carve((size_t)N_NODES * 96 * 4);
  float*  s_acc = (float*)carve((size_t)N_NODES * 128 * 4);
  float*  v_acc = (float*)carve((size_t)N_NODES * 96 * 4);
  float*  vv2   = (float*)carve((size_t)N_NODES * 96 * 4);
  bf16_t* sinb  = (bf16_t*)carve((size_t)N_NODES * 160 * 2);
  float*  deg   = (float*)carve((size_t)N_NODES * 4);
  float*  gcnt  = (float*)carve(G_GRAPH * 4);
  float*  gsum  = (float*)carve(G_GRAPH * 4);
  bf16_t* We1T  = (bf16_t*)carve((size_t)128 * 320 * 2);
  bf16_t* We2T  = (bf16_t*)carve((size_t)192 * 128 * 2);
  bf16_t* WsT   = (bf16_t*)carve((size_t)128 * 128 * 2);
  bf16_t* Wu1T  = (bf16_t*)carve((size_t)160 * 160 * 2);
  bf16_t* Wu2T  = (bf16_t*)carve((size_t)160 * 128 * 2);

  hipMemcpyAsync(s_cur, s_in, (size_t)N_NODES * 128 * 4, hipMemcpyDeviceToDevice, stream);
  hipMemcpyAsync(v_cur, v_in, (size_t)N_NODES * 96 * 4, hipMemcpyDeviceToDevice, stream);
  hipMemsetAsync(deg, 0, (size_t)N_NODES * 4, stream);
  hipMemsetAsync(gcnt, 0, G_GRAPH * 4, stream);
  k_counts<<<(E_EDGES + 255) / 256, 256, 0, stream>>>(dstI, deg, batch, gcnt, E_EDGES, N_NODES);

  int pi = 9;  // params flattened after the 9 array inputs (JAX pytree: dict keys sorted)
  for (int layer = 0; layer < 5; ++layer) {
    const int has_vec = (layer > 0) ? 1 : 0;
    const int use_mlp = (layer < 4) ? 1 : 0;
    const int nout    = has_vec ? 192 : 160;   // We2 output width
    const int u1out   = use_mlp ? 128 : 160;   // Wu1 output width

    const float* We1  = (const float*)d_in[pi++];
    const float* We2  = (const float*)d_in[pi++];
    const float* Ws   = (const float*)d_in[pi++];
    const float* Wu1  = (const float*)d_in[pi++];
    const float* Wu2  = use_mlp ? (const float*)d_in[pi++] : nullptr;
    const float* Wv0  = (const float*)d_in[pi++];
    const float* Wvec = has_vec ? (const float*)d_in[pi++] : nullptr;
    const float* be1  = (const float*)d_in[pi++];
    const float* be2  = (const float*)d_in[pi++];
    const float* bs   = (const float*)d_in[pi++];
    const float* bu1  = (const float*)d_in[pi++];
    const float* bu2  = use_mlp ? (const float*)d_in[pi++] : nullptr;
    const float* ln_b = (const float*)d_in[pi++];
    const float* ln_w = (const float*)d_in[pi++];

    hipMemsetAsync(gsum, 0, G_GRAPH * 4, stream);
    hipMemsetAsync(s_acc, 0, (size_t)N_NODES * 128 * 4, stream);
    hipMemsetAsync(v_acc, 0, (size_t)N_NODES * 96 * 4, stream);

    k_vnorm<<<N_NODES / 4, 128, 0, stream>>>(v_cur, batch, gsum);

    k_w2bf<<<128, 128, 0, stream>>>(We1, We1T, 290, 128, 320);
    k_w2bf<<<nout, 128, 0, stream>>>(We2, We2T, 128, nout, 128);
    k_w2bf<<<128, 128, 0, stream>>>(Ws, WsT, 128, 128, 128);
    k_w2bf<<<u1out, 128, 0, stream>>>(Wu1, Wu1T, 160, u1out, 160);
    if (use_mlp) k_w2bf<<<160, 128, 0, stream>>>(Wu2, Wu2T, 128, 160, 128);

    k_node_prep<<<N_NODES / 4, 128, 0, stream>>>(s_cur, v_cur, s_bf, vbm, batch, gsum, gcnt,
                                                 ln_w, ln_b, Wvec, has_vec);

    k_sb<<<N_NODES / 64, 128, 0, stream>>>(s_bf, WsT, bs, sb);

    if (has_vec)
      k_edge<12><<<E_EDGES / EB, 128, 0, stream>>>(s_bf, sb, vbm, srcI, dstI, dA, aA, eA, rA,
                                                   We1T, be1, We2T, be2, s_acc, v_acc);
    else
      k_edge<10><<<E_EDGES / EB, 128, 0, stream>>>(s_bf, sb, vbm, srcI, dstI, dA, aA, eA, rA,
                                                   We1T, be1, We2T, be2, s_acc, v_acc);

    k_update<<<N_NODES / 4, 128, 0, stream>>>(s_cur, v_cur, s_acc, v_acc, deg, Wv0, vv2, sinb);

    if (use_mlp)
      k_upmlp<<<N_NODES / 64, 128, 0, stream>>>(sinb, Wu1T, bu1, Wu2T, bu2, s_cur, v_cur, vv2);
    else
      k_uplast<<<N_NODES / 64, 128, 0, stream>>>(sinb, Wu1T, bu1, s_cur, v_cur, vv2);
  }

  hipMemcpyAsync(d_out, s_cur, (size_t)N_NODES * 128 * 4, hipMemcpyDeviceToDevice, stream);
  hipMemcpyAsync((float*)d_out + (size_t)N_NODES * 128, v_cur,
                 (size_t)N_NODES * 96 * 4, hipMemcpyDeviceToDevice, stream);
}
